// CrossModalAttention_88725434401067
// MI455X (gfx1250) — matrix operand order repaired and checked by reading, not compiled, not run on hardware
//
#include <hip/hip_runtime.h>

// ---------------------------------------------------------------------------
// CrossModalAttention for MI455X (gfx1250, wave32, WMMA f32<-f16,
// async global->LDS staging, DPP row reductions)
// B=16, T=512, P=576, D=1024, H=16, HD=64
// ---------------------------------------------------------------------------

#define SCALE_ 0.125f   // 1/sqrt(64)

typedef __attribute__((ext_vector_type(16))) _Float16 v16h;
typedef __attribute__((ext_vector_type(8)))  float    v8f;

union AF  { v16h v; unsigned u[8]; _Float16 h[16]; };
union H16 { uint4 q[2]; _Float16 h[16]; };

// A-matrix (16x32 f16) element-pair base K for VGPR j, lane-group g
__device__ __forceinline__ int a_pair_base(int j, int g) {
    return 2 * j + 16 * g;
}

// ---- CDNA5 async global->LDS copy (ASYNCcnt path) -------------------------
__device__ __forceinline__ void async_b128(unsigned lds_addr, const _Float16* g) {
    asm volatile("global_load_async_to_lds_b128 %0, %1, off"
                 :: "v"(lds_addr), "v"(g) : "memory");
}
__device__ __forceinline__ void wait_async0() {
    asm volatile("s_wait_asynccnt 0x0" ::: "memory");
}

// ---- DPP16 row (16-lane) rotate reductions --------------------------------
template <int ROR>
__device__ __forceinline__ float dpp_ror_f(float x) {
    return __int_as_float(__builtin_amdgcn_update_dpp(
        0, __float_as_int(x), 0x120 | ROR, 0xf, 0xf, true));
}
__device__ __forceinline__ float row16_max(float x) {
    x = fmaxf(x, dpp_ror_f<1>(x));
    x = fmaxf(x, dpp_ror_f<2>(x));
    x = fmaxf(x, dpp_ror_f<4>(x));
    x = fmaxf(x, dpp_ror_f<8>(x));
    return x;
}
__device__ __forceinline__ float row16_sum(float x) {
    x += dpp_ror_f<1>(x);
    x += dpp_ror_f<2>(x);
    x += dpp_ror_f<4>(x);
    x += dpp_ror_f<8>(x);
    return x;
}

// ---------------------------------------------------------------------------
// fp32 -> f16 conversion, 8 elements/thread (n must be multiple of 2048)
// ---------------------------------------------------------------------------
__global__ __launch_bounds__(256)
void cvt_f16_kernel(const float* __restrict__ in, _Float16* __restrict__ out)
{
    const size_t i = ((size_t)blockIdx.x * 256 + threadIdx.x) * 8;
    const float4 f0 = ((const float4*)(in + i))[0];
    const float4 f1 = ((const float4*)(in + i))[1];
    H16 o;
    o.h[0] = (_Float16)f0.x; o.h[1] = (_Float16)f0.y;
    o.h[2] = (_Float16)f0.z; o.h[3] = (_Float16)f0.w;
    o.h[4] = (_Float16)f1.x; o.h[5] = (_Float16)f1.y;
    o.h[6] = (_Float16)f1.z; o.h[7] = (_Float16)f1.w;
    ((uint4*)(out + i))[0] = o.q[0];
}

// ---------------------------------------------------------------------------
// GEMM:  C[m,n] = sum_k A[m,k] * W[n,k] + bias[n]   (A,W f16; K = N = 1024)
// 256 threads (8 waves), tile 128(M) x 64(N), BK=32, double-buffered async LDS.
// OUT_MODE 0: f16 scatter to [B,H,S,64]   OUT_MODE 2: fp32 to [M,1024]
// ---------------------------------------------------------------------------
__device__ __forceinline__ void stage_gemm_tile(
    _Float16* asbuf, _Float16* bsbuf,
    const _Float16* __restrict__ A, const _Float16* __restrict__ Wh,
    int m0, int n0, int k0, int t)
{
    const int ar = t >> 1, ak = (t & 1) << 4;   // A: 128 rows x 32, 32B/thread
    const int bn = t >> 2, bk = (t & 3) << 3;   // B: 64 rows x 32, 16B/thread
    const unsigned la = (unsigned)(uintptr_t)(asbuf + ar * 32 + ak);
    const _Float16* ga = A + (size_t)(m0 + ar) * 1024 + k0 + ak;
    async_b128(la, ga);
    async_b128(la + 16u, ga + 8);
    const unsigned lb = (unsigned)(uintptr_t)(bsbuf + bn * 32 + bk);
    async_b128(lb, Wh + (size_t)(n0 + bn) * 1024 + k0 + bk);
}

template <int OUT_MODE>
__global__ __launch_bounds__(256)
void gemm_nt_f16_kernel(const _Float16* __restrict__ A,
                        const _Float16* __restrict__ Wh,
                        const float* __restrict__ bias,
                        void* __restrict__ Outp,
                        int M, int S)
{
    __shared__ _Float16 As[2][128 * 32];
    __shared__ _Float16 Bs[2][64 * 32];

    const int t    = threadIdx.x;
    const int wid  = t >> 5;
    const int lane = t & 31;
    const int ml   = lane & 15;
    const int g    = lane >> 4;
    const int m0   = blockIdx.x * 128;
    const int n0   = blockIdx.y * 64;

    v8f acc[4] = {};

    stage_gemm_tile(As[0], Bs[0], A, Wh, m0, n0, 0, t);
    wait_async0();
    __syncthreads();

    int cur = 0;
    for (int k0 = 0; k0 < 1024; k0 += 32) {
        if (k0 + 32 < 1024)
            stage_gemm_tile(As[cur ^ 1], Bs[cur ^ 1], A, Wh, m0, n0, k0 + 32, t);

        // A fragment (16x32) for this wave's 16 rows
        AF a;
#pragma unroll
        for (int j = 0; j < 8; ++j)
            a.u[j] = *(const unsigned*)(As[cur] + (wid * 16 + ml) * 32 + a_pair_base(j, g));

#pragma unroll
        for (int ns = 0; ns < 4; ++ns) {
            AF b;
#pragma unroll
            for (int j = 0; j < 8; ++j)
                b.u[j] = *(const unsigned*)(Bs[cur] + (ns * 16 + ml) * 32 + 2 * j + 16 * g);
            acc[ns] = __builtin_amdgcn_wmma_f32_16x16x32_f16(
                false, a.v, false, b.v, (short)0, acc[ns], false, false);
        }

        wait_async0();
        __syncthreads();
        cur ^= 1;
    }

    // ---- epilogue ----
#pragma unroll
    for (int ns = 0; ns < 4; ++ns) {
        const int   col = n0 + ns * 16 + ml;
        const float bv  = bias[col];
#pragma unroll
        for (int r = 0; r < 8; ++r) {
            const int   row = m0 + wid * 16 + r + 8 * g;
            const float val = acc[ns][r] + bv;
            if (OUT_MODE == 0) {
                const int bb = row / S, ss = row - bb * S;
                const int hh = col >> 6, dd = col & 63;
                ((_Float16*)Outp)[(((size_t)(bb * 16 + hh) * S + ss) << 6) + dd] =
                    (_Float16)val;
            } else {
                ((float*)Outp)[(size_t)row * 1024 + col] = val;
            }
        }
    }
}

// ---------------------------------------------------------------------------
// Flash attention: one WG = 4 waves = 64 query rows of one (b,h).
// Q/K/V: f16 [B,H,S,64].  Out: f16 merged [B,Sq,1024].
// ---------------------------------------------------------------------------
__global__ __launch_bounds__(128)
void attn_kernel(const _Float16* __restrict__ Q,
                 const _Float16* __restrict__ Km,
                 const _Float16* __restrict__ Vm,
                 _Float16* __restrict__ Out,
                 const unsigned char* __restrict__ mask,
                 int Sq, int Sk)
{
    __shared__ _Float16 Ks[32 * 64];       // [key][d]
    __shared__ _Float16 Vt[64 * 32];       // [d][key]
    __shared__ _Float16 Pr[4][16 * 32];    // per-wave probs [m][key]

    const int t    = threadIdx.x;
    const int wid  = t >> 5;
    const int lane = t & 31;
    const int ml   = lane & 15;
    const int g    = lane >> 4;
    const int bh   = blockIdx.y;
    const int b    = bh >> 4, h = bh & 15;
    const int q0   = blockIdx.x * 64 + wid * 16;

    // Q fragments (d = 0..31, 32..63), held in registers for the whole pass
    AF a0, a1;
    {
        const _Float16* qrow = Q + ((size_t)bh * Sq + q0 + ml) * 64;
#pragma unroll
        for (int j = 0; j < 8; ++j) {
            const int kb = a_pair_base(j, g);
            a0.u[j] = *(const unsigned*)(qrow + kb);
            a1.u[j] = *(const unsigned*)(qrow + kb + 32);
        }
    }

    float rmax[8], rsum[8];
#pragma unroll
    for (int r = 0; r < 8; ++r) { rmax[r] = -1e30f; rsum[r] = 0.f; }
    v8f O[4] = {};

    const unsigned char* mrow = mask ? (mask + (size_t)b * Sk) : nullptr;

    const int srow = t >> 2;             // staging: key 0..31
    const int sseg = (t & 3) << 4;       // staging: d segment of 16

    for (int kb0 = 0; kb0 < Sk; kb0 += 32) {
        // ---- stage K tile asynchronously, V tile transposed via registers ----
        {
            const size_t base = ((size_t)bh * Sk + kb0 + srow) * 64 + sseg;
            const unsigned lk = (unsigned)(uintptr_t)(Ks + srow * 64 + sseg);
            async_b128(lk, Km + base);
            async_b128(lk + 16u, Km + base + 8);

            H16 tmp;
            const uint4* vs = (const uint4*)(Vm + base);
            tmp.q[0] = vs[0];
            tmp.q[1] = vs[1];
#pragma unroll
            for (int i = 0; i < 16; ++i)
                Vt[(sseg + i) * 32 + srow] = tmp.h[i];
        }
        wait_async0();
        __syncthreads();

        // ---- scores: S = Q . K^T  (two 16-key tiles, K-dim = 64) ----
        v8f s0 = {}, s1 = {};
        {
            AF bk0, bk1;
#pragma unroll
            for (int j = 0; j < 8; ++j) {
                const int dof = 2 * j + 16 * g;
                bk0.u[j] = *(const unsigned*)(Ks + ml * 64 + dof);
                bk1.u[j] = *(const unsigned*)(Ks + (16 + ml) * 64 + dof);
            }
            s0 = __builtin_amdgcn_wmma_f32_16x16x32_f16(false, a0.v, false, bk0.v,
                                                        (short)0, s0, false, false);
            s1 = __builtin_amdgcn_wmma_f32_16x16x32_f16(false, a0.v, false, bk1.v,
                                                        (short)0, s1, false, false);
#pragma unroll
            for (int j = 0; j < 8; ++j) {
                const int dof = 32 + 2 * j + 16 * g;
                bk0.u[j] = *(const unsigned*)(Ks + ml * 64 + dof);
                bk1.u[j] = *(const unsigned*)(Ks + (16 + ml) * 64 + dof);
            }
            s0 = __builtin_amdgcn_wmma_f32_16x16x32_f16(false, a1.v, false, bk0.v,
                                                        (short)0, s0, false, false);
            s1 = __builtin_amdgcn_wmma_f32_16x16x32_f16(false, a1.v, false, bk1.v,
                                                        (short)0, s1, false, false);
        }

        // ---- online softmax update (DPP row reductions) ----
        bool k0ok = true, k1ok = true;
        if (mrow) {
            k0ok = mrow[kb0 + ml] != 0;
            k1ok = mrow[kb0 + 16 + ml] != 0;
        }
        float corr[8];
#pragma unroll
        for (int r = 0; r < 8; ++r) {
            const float v0 = k0ok ? s0[r] * SCALE_ : -1e30f;
            const float v1 = k1ok ? s1[r] * SCALE_ : -1e30f;
            const float tm = row16_max(fmaxf(v0, v1));
            const float nm = fmaxf(rmax[r], tm);
            const float c  = __expf(rmax[r] - nm);
            rmax[r] = nm;
            const float p0 = __expf(v0 - nm);
            const float p1 = __expf(v1 - nm);
            s0[r] = p0;
            s1[r] = p1;
            rsum[r] = rsum[r] * c + row16_sum(p0 + p1);
            corr[r] = c;
        }
#pragma unroll
        for (int dt = 0; dt < 4; ++dt)
#pragma unroll
            for (int r = 0; r < 8; ++r)
                O[dt][r] *= corr[r];

        // ---- probs -> per-wave LDS -> A fragment (lane transpose) ----
        _Float16* pw = &Pr[wid][0];
#pragma unroll
        for (int r = 0; r < 8; ++r) {
            const int m = r + 8 * g;
            pw[m * 32 + ml]      = (_Float16)s0[r];
            pw[m * 32 + 16 + ml] = (_Float16)s1[r];
        }
        AF pa;
#pragma unroll
        for (int j = 0; j < 8; ++j)
            pa.u[j] = *(const unsigned*)(pw + ml * 32 + a_pair_base(j, g));

        // ---- O += P . V  (4 d-tiles of 16) ----
#pragma unroll
        for (int dt = 0; dt < 4; ++dt) {
            AF bv;
#pragma unroll
            for (int j = 0; j < 8; ++j)
                bv.u[j] = *(const unsigned*)(Vt + (dt * 16 + ml) * 32 + 2 * j + 16 * g);
            O[dt] = __builtin_amdgcn_wmma_f32_16x16x32_f16(
                false, pa.v, false, bv.v, (short)0, O[dt], false, false);
        }
        __syncthreads();
    }

    // ---- normalize and write merged-head f16 output [B,Sq,1024] ----
#pragma unroll
    for (int r = 0; r < 8; ++r) rsum[r] = 1.f / rsum[r];
#pragma unroll
    for (int dt = 0; dt < 4; ++dt) {
        const int col = h * 64 + dt * 16 + ml;
#pragma unroll
        for (int r = 0; r < 8; ++r) {
            const int s = q0 + r + 8 * g;
            Out[((size_t)b * Sq + s) * 1024 + col] = (_Float16)(O[dt][r] * rsum[r]);
        }
    }
}

// ---------------------------------------------------------------------------
// Launch
// ---------------------------------------------------------------------------
extern "C" void kernel_launch(void* const* d_in, const int* in_sizes, int n_in,
                              void* d_out, int out_size, void* d_ws, size_t ws_size,
                              hipStream_t stream)
{
    (void)in_sizes; (void)n_in; (void)out_size; (void)ws_size;

    const float*         text  = (const float*)d_in[0];
    const float*         image = (const float*)d_in[1];
    const unsigned char* mask  = (const unsigned char*)d_in[2];
    const float* w_qt = (const float*)d_in[3];  const float* b_qt = (const float*)d_in[4];
    const float* w_ki = (const float*)d_in[5];  const float* b_ki = (const float*)d_in[6];
    const float* w_vi = (const float*)d_in[7];  const float* b_vi = (const float*)d_in[8];
    const float* w_qi = (const float*)d_in[9];  const float* b_qi = (const float*)d_in[10];
    const float* w_kt = (const float*)d_in[11]; const float* b_kt = (const float*)d_in[12];
    const float* w_vt = (const float*)d_in[13]; const float* b_vt = (const float*)d_in[14];
    const float* w_ot = (const float*)d_in[15]; const float* b_ot = (const float*)d_in[16];
    const float* w_oi = (const float*)d_in[17]; const float* b_oi = (const float*)d_in[18];

    _Float16* ws = (_Float16*)d_ws;
    const size_t TSZ = (size_t)16 * 16 * 512 * 64;  // 8,388,608 halves
    const size_t ISZ = (size_t)16 * 16 * 576 * 64;  // 9,437,184 halves
    const size_t WSZ = (size_t)1024 * 1024;
    _Float16* qT = ws;
    _Float16* kT = qT + TSZ;
    _Float16* vT = kT + TSZ;
    _Float16* qI = vT + TSZ;
    _Float16* kI = qI + ISZ;
    _Float16* vI = kI + ISZ;
    _Float16* aT = vI + ISZ;      // attended text  [16,512,1024] f16
    _Float16* aI = aT + TSZ;      // attended image [16,576,1024] f16
    _Float16* tX = aI + ISZ;      // text feats  f16
    _Float16* iX = tX + TSZ;      // image feats f16
    _Float16* hw[8];              // f16 weights: qt,ki,vi,qi,kt,vt,ot,oi
    hw[0] = iX + ISZ;
    for (int i = 1; i < 8; ++i) hw[i] = hw[i - 1] + WSZ;

    // ---- one-shot fp32 -> f16 conversions ----
    cvt_f16_kernel<<<dim3(4096), 256, 0, stream>>>(text,  tX);
    cvt_f16_kernel<<<dim3(4608), 256, 0, stream>>>(image, iX);
    const float* wsrc[8] = { w_qt, w_ki, w_vi, w_qi, w_kt, w_vt, w_ot, w_oi };
    for (int i = 0; i < 8; ++i)
        cvt_f16_kernel<<<dim3(512), 256, 0, stream>>>(wsrc[i], hw[i]);

    // ---- QKV projections (f16 in, f16 head-layout out) ----
    gemm_nt_f16_kernel<0><<<dim3(64, 16), 256, 0, stream>>>(tX, hw[0], b_qt, qT, 8192, 512);
    gemm_nt_f16_kernel<0><<<dim3(64, 16), 256, 0, stream>>>(tX, hw[4], b_kt, kT, 8192, 512);
    gemm_nt_f16_kernel<0><<<dim3(64, 16), 256, 0, stream>>>(tX, hw[5], b_vt, vT, 8192, 512);
    gemm_nt_f16_kernel<0><<<dim3(72, 16), 256, 0, stream>>>(iX, hw[3], b_qi, qI, 9216, 576);
    gemm_nt_f16_kernel<0><<<dim3(72, 16), 256, 0, stream>>>(iX, hw[1], b_ki, kI, 9216, 576);
    gemm_nt_f16_kernel<0><<<dim3(72, 16), 256, 0, stream>>>(iX, hw[2], b_vi, vI, 9216, 576);

    // ---- cross attention (flash, online softmax) ----
    attn_kernel<<<dim3(8, 256), 128, 0, stream>>>(qT, kI, vI, aT, nullptr, 512, 576);
    attn_kernel<<<dim3(9, 256), 128, 0, stream>>>(qI, kT, vT, aI, mask,    576, 512);

    // ---- output projections (f16 in, fp32 out -> d_out) ----
    float* outT = (float*)d_out;
    float* outI = outT + (size_t)8192 * 1024;
    gemm_nt_f16_kernel<2><<<dim3(64, 16), 256, 0, stream>>>(aT, hw[6], b_ot, outT, 8192, 512);
    gemm_nt_f16_kernel<2><<<dim3(72, 16), 256, 0, stream>>>(aI, hw[7], b_oi, outI, 9216, 576);
}
